// BidirectionalMultiheadSelfAttention_14285061226815
// MI455X (gfx1250) — matrix operand order repaired and checked by reading, not compiled, not run on hardware
//
#include <hip/hip_runtime.h>
#include <stdint.h>

// Fixed problem shape from the reference.
#define T_DIM 1024
#define B_DIM 8
#define E_DIM 512
#define H_DIM 8
#define D_DIM 64
#define S_DIM 2048   // 2*T

// Use GLOBAL_LOAD_ASYNC_TO_LDS_B128 (+ s_wait_asynccnt) for the GEMM weight
// staging. Flip to 0 if the assembler rejects the mnemonic.
#define USE_ASYNC_LDS 1

typedef __attribute__((ext_vector_type(16))) __bf16 v16bf;
typedef __attribute__((ext_vector_type(8)))  float  v8f;

union ABRegs { uint4 q[2]; v16bf v; };

__device__ __forceinline__ unsigned short f2bf(float f) {
  unsigned int u = __float_as_uint(f);
  if ((u & 0x7F800000u) != 0x7F800000u)          // keep inf/nan bit patterns
    u += 0x7FFFu + ((u >> 16) & 1u);             // round-to-nearest-even
  return (unsigned short)(u >> 16);
}

// A-operand 16x32 bf16 tile (wave32 layout: lane%16 = M row, lane/16 selects
// K-halves; VGPR v holds bf16 pair). Source row-major [rows, ld] bf16.
// Per-lane dword pattern = two contiguous 16B runs -> 2x b128 loads.
__device__ __forceinline__ v16bf load_a16x32(const unsigned int* base, int row0,
                                             int ld, int kbase, int lane) {
  int m = row0 + (lane & 15);
  int g = lane >> 4;
  const uint4* p = (const uint4*)(base + ((m * ld + kbase) >> 1));
  ABRegs u;
  u.q[0] = p[g];         // K = kbase + g*8 + {0..7}
  u.q[1] = p[g + 2];     // K = kbase + 16 + g*8 + {0..7}
  return u.v;
}

// B-operand 32x16 bf16 tile (lane%16 = N col, lane halves select K 8-runs
// like the A operand). Source row-major [cols, ld] (i.e. B^T),
// so K-pairs are contiguous -> 2x b128 loads (global or ds).
__device__ __forceinline__ v16bf load_b32x16(const unsigned int* base, int col0,
                                             int ld, int kbase, int lane) {
  int n = col0 + (lane & 15);
  int g = lane >> 4;
  const uint4* p = (const uint4*)(base + ((n * ld + kbase) >> 1));
  ABRegs u;
  u.q[0] = p[g];         // K = kbase + g*8 + {0..7}
  u.q[1] = p[g + 2];     // K = kbase + 16 + g*8 + {0..7}
  return u.v;
}

__device__ __forceinline__ v8f wmma_bf16(v16bf a, v16bf b, v8f c) {
  return __builtin_amdgcn_wmma_f32_16x16x32_bf16(false, a, false, b,
                                                 (short)0, c, false, false);
}

// ---------------------------------------------------------------------------
// Prep: q_in = shift(fwd)+shift(bwd) -> bf16 ; kv = concat(fwd,bwd) -> bf16
// ---------------------------------------------------------------------------
__global__ void prep_inputs_kernel(const float* __restrict__ fwd,
                                   const float* __restrict__ bwd,
                                   unsigned short* __restrict__ qin,
                                   unsigned short* __restrict__ kv) {
  const int total = T_DIM * B_DIM * E_DIM;
  int idx = blockIdx.x * blockDim.x + threadIdx.x;
  if (idx >= total) return;
  const int BE = B_DIM * E_DIM;
  int t = idx / BE;
  float q = 0.f;
  if (t > 0)          q += fwd[idx - BE];
  if (t < T_DIM - 1)  q += bwd[idx + BE];
  qin[idx] = f2bf(q);
  kv[idx]          = f2bf(fwd[idx]);
  kv[idx + total]  = f2bf(bwd[idx]);
}

__global__ void conv_weights_kernel(const float* __restrict__ ipw,
                                    const float* __restrict__ ow,
                                    unsigned short* __restrict__ wqkv,
                                    unsigned short* __restrict__ wo) {
  int idx = blockIdx.x * blockDim.x + threadIdx.x;
  const int n3 = 3 * E_DIM * E_DIM;
  if (idx < n3) wqkv[idx] = f2bf(ipw[idx]);
  if (idx < E_DIM * E_DIM) wo[idx] = f2bf(ow[idx]);
}

// ---------------------------------------------------------------------------
// Generic NT GEMM: C[M,N] = A[M,K] * W[N,K]^T + bias.
// Block = 8 waves = 128 rows x one 64-col n-tile. The 64x512 bf16 weight slab
// (64KB) is staged once into LDS (async DMA, ASYNCcnt) and shared by all
// 8 waves; B-operand tiles then come from LDS (ds_load_b128), A from global.
// mode 0: Q -> bf16 [B,H,T,D]     mode 1: K -> bf16 [B,H,S,D]
// mode 2: V -> bf16 [B,H,D,S]     mode 3: f32 row-major (final out proj)
// ---------------------------------------------------------------------------
__global__ __launch_bounds__(256)
void gemm_nt_kernel(const unsigned short* __restrict__ A,
                    const unsigned short* __restrict__ W,
                    const float* __restrict__ bias,
                    int M, int mode,
                    unsigned short* __restrict__ obf,
                    float* __restrict__ ofp) {
  const int K = E_DIM, N = E_DIM;
  __shared__ unsigned short wslab[64 * E_DIM];     // 64 KB
  const int tid  = threadIdx.x;
  const int lane = tid & 31;
  const int wid  = tid >> 5;
  const int ntile = blockIdx.x & 7;                // 8 n-tiles of 64 cols
  const int mtile = (blockIdx.x >> 3) * 8 + wid;   // 8 m-tiles per block

  // ---- stage weight slab: rows [ntile*64, +64) of W, contiguous 64KB
  {
    const uint4* src = (const uint4*)(W + (size_t)ntile * 64 * K);
#if USE_ASYNC_LDS
    unsigned int lds0 = (unsigned int)(size_t)&wslab[0]; // wave-rel LDS offset
    for (int i = tid; i < (64 * K) / 8; i += 256) {
      unsigned int ldsoff = lds0 + i * 16;
      const uint4* gp = src + i;
      asm volatile("global_load_async_to_lds_b128 %0, %1, off"
                   :: "v"(ldsoff), "v"(gp)
                   : "memory");
    }
    asm volatile("s_wait_asynccnt 0x0" ::: "memory");
#else
    uint4* dst = (uint4*)wslab;
    for (int i = tid; i < (64 * K) / 8; i += 256) dst[i] = src[i];
#endif
  }
  __syncthreads();

  const unsigned int* Ad = (const unsigned int*)A;
  const unsigned int* Wl = (const unsigned int*)wslab;   // LDS, ld = K

  v8f acc[4];
  #pragma unroll
  for (int nc = 0; nc < 4; ++nc)
    acc[nc] = (v8f){0.f,0.f,0.f,0.f,0.f,0.f,0.f,0.f};

  for (int kk = 0; kk < K; kk += 32) {
    v16bf a = load_a16x32(Ad, mtile * 16, K, kk, lane);
    if (kk + 32 < K)
      __builtin_prefetch(Ad + (((mtile * 16 + (lane & 15)) * K + kk + 32) >> 1), 0, 1);
    #pragma unroll
    for (int nc = 0; nc < 4; ++nc) {
      v16bf bt = load_b32x16(Wl, nc * 16, K, kk, lane);  // from LDS
      acc[nc] = wmma_bf16(a, bt, acc[nc]);
    }
  }

  const int g = lane >> 4, nl = lane & 15;
  #pragma unroll
  for (int nc = 0; nc < 4; ++nc) {
    #pragma unroll
    for (int r = 0; r < 8; ++r) {
      int row = mtile * 16 + r + 8 * g;            // row = t*B + b
      int col = ntile * 64 + nc * 16 + nl;         // col = h*64 + d
      float val = acc[nc][r] + bias[col];
      if (mode == 3) {
        ofp[(size_t)row * N + col] = val;
      } else {
        int bb = row & (B_DIM - 1), tt = row >> 3;
        int h = col >> 6, d = col & 63;
        size_t bh = (size_t)(bb * H_DIM + h);
        if (mode == 0)      obf[(bh * T_DIM + tt) * D_DIM + d] = f2bf(val);
        else if (mode == 1) obf[(bh * S_DIM + tt) * D_DIM + d] = f2bf(val);
        else                obf[(bh * D_DIM + d) * S_DIM + tt] = f2bf(val);
      }
    }
  }
}

// ---------------------------------------------------------------------------
// Fused attention. Block = (b, 16 query rows), 8 waves, each wave owns a
// 256-wide slab of s. Loops all 8 heads; avg_weights accumulates across heads
// in a 128KB LDS tile (CDNA5: 320KB/WGP) -> single global write, no atomics.
// ---------------------------------------------------------------------------
__global__ __launch_bounds__(256)
void attention_kernel(const unsigned short* __restrict__ Qb,
                      const unsigned short* __restrict__ Kb,
                      const unsigned short* __restrict__ Vtb,
                      const unsigned char* __restrict__ kpm,
                      unsigned short* __restrict__ attn,
                      float* __restrict__ avg_out) {
  extern __shared__ unsigned char smem_raw[];
  float* avg  = (float*)smem_raw;                       // 16*2048 f32 = 128KB
  float* red  = avg + 16 * S_DIM;                       // 8 waves * 16 rows
  float* ared = red + 128;                              // 8*16*64 f32 = 32KB
  unsigned short* pst = (unsigned short*)(ared + 8 * 16 * 64); // 8*16*32 bf16

  const int tid  = threadIdx.x;
  const int lane = tid & 31, w = tid >> 5;
  const int g = lane >> 4, nl = lane & 15;
  const int b  = blockIdx.x >> 6;                       // 64 tiles per batch
  const int t0 = (blockIdx.x & 63) << 4;
  const int sbase = w * 256;

  for (int i = tid; i < 16 * S_DIM; i += 256) avg[i] = 0.f;

  // padding-mask bits for this lane's 16 columns per subtile (head-invariant)
  unsigned int padbits = 0;
  #pragma unroll
  for (int sc = 0; sc < 16; ++sc) {
    int ss = sbase + sc * 16 + nl;
    if (kpm[b * T_DIM + (ss & (T_DIM - 1))]) padbits |= 1u << sc;
  }
  __syncthreads();

  const unsigned int* Qd = (const unsigned int*)Qb;
  const unsigned int* Kd = (const unsigned int*)Kb;
  const unsigned int* Vd = (const unsigned int*)Vtb;
  const float NEGINF = -__builtin_inff();

  for (int h = 0; h < H_DIM; ++h) {
    const unsigned int* Qh = Qd + ((size_t)(b * H_DIM + h) * T_DIM * D_DIM >> 1);
    const unsigned int* Kh = Kd + ((size_t)(b * H_DIM + h) * S_DIM * D_DIM >> 1);
    const unsigned int* Vh = Vd + ((size_t)(b * H_DIM + h) * D_DIM * S_DIM >> 1);

    v16bf qa0 = load_a16x32(Qh, t0, D_DIM, 0, lane);
    v16bf qa1 = load_a16x32(Qh, t0, D_DIM, 32, lane);

    // ---- scores = Q K^T over this wave's 16x256 slab (no 1/sqrt(D) in ref)
    v8f sacc[16];
    #pragma unroll
    for (int sc = 0; sc < 16; ++sc) {
      int st = sbase + sc * 16;
      v16bf kb0 = load_b32x16(Kh, st, D_DIM, 0, lane);
      v16bf kb1 = load_b32x16(Kh, st, D_DIM, 32, lane);
      v8f c = (v8f){0.f,0.f,0.f,0.f,0.f,0.f,0.f,0.f};
      c = wmma_bf16(qa0, kb0, c);
      c = wmma_bf16(qa1, kb1, c);
      sacc[sc] = c;
    }

    // ---- mask + row max (allowed = s < t || s > t+T, minus key padding)
    float rmax[8];
    #pragma unroll
    for (int r = 0; r < 8; ++r) rmax[r] = NEGINF;
    #pragma unroll
    for (int sc = 0; sc < 16; ++sc) {
      int ss = sbase + sc * 16 + nl;
      bool pad = (padbits >> sc) & 1u;
      #pragma unroll
      for (int r = 0; r < 8; ++r) {
        int tt = t0 + r + 8 * g;
        float x = sacc[sc][r];
        bool ok = ((ss < tt) || (ss > tt + T_DIM)) && !pad;
        x = ok ? x : NEGINF;
        sacc[sc][r] = x;
        rmax[r] = fmaxf(rmax[r], x);
      }
    }
    #pragma unroll
    for (int off = 1; off < 16; off <<= 1)
      #pragma unroll
      for (int r = 0; r < 8; ++r)
        rmax[r] = fmaxf(rmax[r], __shfl_xor(rmax[r], off, 32));
    if (nl == 0) {
      #pragma unroll
      for (int r = 0; r < 8; ++r) red[w * 16 + g * 8 + r] = rmax[r];
    }
    __syncthreads();
    float gmax[8];
    #pragma unroll
    for (int r = 0; r < 8; ++r) {
      float m = NEGINF;
      for (int ww = 0; ww < 8; ++ww) m = fmaxf(m, red[ww * 16 + g * 8 + r]);
      gmax[r] = m;
    }
    __syncthreads();

    // ---- exp + row sum
    float rsum[8];
    #pragma unroll
    for (int r = 0; r < 8; ++r) rsum[r] = 0.f;
    #pragma unroll
    for (int sc = 0; sc < 16; ++sc) {
      #pragma unroll
      for (int r = 0; r < 8; ++r) {
        float e = __expf(sacc[sc][r] - gmax[r]);
        sacc[sc][r] = e;
        rsum[r] += e;
      }
    }
    #pragma unroll
    for (int off = 1; off < 16; off <<= 1)
      #pragma unroll
      for (int r = 0; r < 8; ++r)
        rsum[r] += __shfl_xor(rsum[r], off, 32);
    if (nl == 0) {
      #pragma unroll
      for (int r = 0; r < 8; ++r) red[w * 16 + g * 8 + r] = rsum[r];
    }
    __syncthreads();
    float inv[8];
    #pragma unroll
    for (int r = 0; r < 8; ++r) {
      float s = 0.f;
      for (int ww = 0; ww < 8; ++ww) s += red[ww * 16 + g * 8 + r];
      inv[r] = (s > 0.f) ? (1.f / s) : 0.f;
    }

    // ---- normalize; accumulate head-mean into LDS avg tile
    #pragma unroll
    for (int sc = 0; sc < 16; ++sc) {
      int col = sbase + sc * 16 + nl;
      #pragma unroll
      for (int r = 0; r < 8; ++r) {
        float p = sacc[sc][r] * inv[r];
        sacc[sc][r] = p;
        avg[(r + 8 * g) * S_DIM + col] += p * (1.f / H_DIM);
      }
    }

    // ---- attn += P V : restage P (C-layout f32 -> A-layout bf16) via LDS
    v8f av[4];
    #pragma unroll
    for (int dc = 0; dc < 4; ++dc)
      av[dc] = (v8f){0.f,0.f,0.f,0.f,0.f,0.f,0.f,0.f};
    unsigned short* pw = pst + w * 512;                 // private 16x32 tile
    #pragma unroll
    for (int ks = 0; ks < 8; ++ks) {
      #pragma unroll
      for (int sub = 0; sub < 2; ++sub) {
        int sc = ks * 2 + sub;
        #pragma unroll
        for (int r = 0; r < 8; ++r)
          pw[(r + 8 * g) * 32 + sub * 16 + nl] = f2bf(sacc[sc][r]);
      }
      v16bf pa = load_a16x32((const unsigned int*)pw, 0, 32, 0, lane);
      #pragma unroll
      for (int dc = 0; dc < 4; ++dc) {
        v16bf vb = load_b32x16(Vh, dc * 16, S_DIM, sbase + ks * 32, lane);
        av[dc] = wmma_bf16(pa, vb, av[dc]);
      }
    }

    // ---- cross-wave reduce attn (each wave has a partial over its s-slab)
    #pragma unroll
    for (int dc = 0; dc < 4; ++dc)
      #pragma unroll
      for (int r = 0; r < 8; ++r)
        ared[(w * 16 + r + 8 * g) * 64 + dc * 16 + nl] = av[dc][r];
    __syncthreads();
    {
      int e0 = tid * 4;                               // 1024 elems / 256 thr
      #pragma unroll
      for (int k = 0; k < 4; ++k) {
        int e = e0 + k, row = e >> 6, col = e & 63;
        float s = 0.f;
        for (int ww = 0; ww < 8; ++ww) s += ared[(ww * 16 + row) * 64 + col];
        attn[((size_t)(t0 + row) * B_DIM + b) * E_DIM + h * D_DIM + col] = f2bf(s);
      }
    }
    __syncthreads();
  }

  // ---- write head-averaged weights: avg_out[b, t0:t0+16, :]
  for (int i = tid; i < 16 * S_DIM; i += 256) {
    int row = i >> 11, col = i & (S_DIM - 1);
    avg_out[(size_t)b * T_DIM * S_DIM + (size_t)(t0 + row) * S_DIM + col] = avg[i];
  }
}

// ---------------------------------------------------------------------------
extern "C" void kernel_launch(void* const* d_in, const int* in_sizes, int n_in,
                              void* d_out, int out_size, void* d_ws, size_t ws_size,
                              hipStream_t stream) {
  const float* fwd_x = (const float*)d_in[0];
  const float* bwd_x = (const float*)d_in[1];
  const unsigned char* kpm = (const unsigned char*)d_in[2]; // jax bool -> 1B
  const float* ipw  = (const float*)d_in[3];
  const float* ipb  = (const float*)d_in[4];
  const float* outw = (const float*)d_in[5];
  const float* outb = (const float*)d_in[6];

  float* out_proj = (float*)d_out;                                  // T*B*E
  float* avg_out  = out_proj + (size_t)T_DIM * B_DIM * E_DIM;       // B*T*S

  // workspace carve-up (all sizes multiples of 256B); total ~74 MB
  unsigned char* ws = (unsigned char*)d_ws;
  size_t cur = 0;
  auto take = [&](size_t bytes) { unsigned char* p = ws + cur; cur += bytes; return p; };
  const size_t TBE = (size_t)T_DIM * B_DIM * E_DIM;
  unsigned short* qin  = (unsigned short*)take(TBE * 2);            // [T,B,E]
  unsigned short* kv   = (unsigned short*)take(2 * TBE * 2);        // [S,B,E]
  unsigned short* wqkv = (unsigned short*)take((size_t)3 * E_DIM * E_DIM * 2);
  unsigned short* wo   = (unsigned short*)take((size_t)E_DIM * E_DIM * 2);
  unsigned short* Qb   = (unsigned short*)take(TBE * 2);            // [B,H,T,D]
  unsigned short* Kb   = (unsigned short*)take(2 * TBE * 2);        // [B,H,S,D]
  unsigned short* Vtb  = (unsigned short*)take(2 * TBE * 2);        // [B,H,D,S]
  unsigned short* attn = (unsigned short*)take(TBE * 2);            // [T,B,E]
  (void)ws_size;

  // 1) elementwise prep -> bf16
  prep_inputs_kernel<<<(int)((TBE + 255) / 256), 256, 0, stream>>>(fwd_x, bwd_x, qin, kv);
  conv_weights_kernel<<<(3 * E_DIM * E_DIM + 255) / 256, 256, 0, stream>>>(ipw, outw, wqkv, wo);

  // 2) QKV projections (WMMA bf16, f32 acc); grid = M/16 blocks (8 waves each)
  const int Mq = T_DIM * B_DIM;        // 8192
  const int Mkv = S_DIM * B_DIM;       // 16384
  gemm_nt_kernel<<<Mq / 16, 256, 0, stream>>>(
      qin, wqkv,                 ipb,             Mq, 0, Qb,  nullptr);
  gemm_nt_kernel<<<Mkv / 16, 256, 0, stream>>>(
      kv,  wqkv + (size_t)E_DIM * E_DIM,     ipb + E_DIM,     Mkv, 1, Kb,  nullptr);
  gemm_nt_kernel<<<Mkv / 16, 256, 0, stream>>>(
      kv,  wqkv + (size_t)2 * E_DIM * E_DIM, ipb + 2 * E_DIM, Mkv, 2, Vtb, nullptr);

  // 3) fused masked attention + softmax + PV + head-mean weights
  const size_t smem = (size_t)16 * S_DIM * 4     // avg tile  (128 KB)
                    + 128 * 4                    // reduction scratch
                    + (size_t)8 * 16 * 64 * 4    // attn cross-wave (32 KB)
                    + (size_t)8 * 16 * 32 * 2;   // P restage (8 KB)
  attention_kernel<<<B_DIM * (T_DIM / 16), 256, smem, stream>>>(
      Qb, Kb, Vtb, kpm, attn, avg_out);

  // 4) output projection -> f32 d_out
  gemm_nt_kernel<<<Mq / 16, 256, 0, stream>>>(
      attn, wo, outb, Mq, 3, nullptr, out_proj);
}
